// GraphAttentionLayer_59760174957313
// MI455X (gfx1250) — matrix operand order repaired and checked by reading, not compiled, not run on hardware
//
#include <hip/hip_runtime.h>
#include <hip/hip_bf16.h>

typedef _Float16 f16;
typedef __attribute__((ext_vector_type(16))) _Float16 v16h;
typedef __attribute__((ext_vector_type(8)))  _Float16 v8h;
typedef __attribute__((ext_vector_type(8)))  float    v8f;
typedef __attribute__((ext_vector_type(4)))  int      v4i;

#define LRELU_SLOPE 0.2f
#define NEG_BIG (-9.0e15f)

#define AS1 __attribute__((address_space(1)))
#define AS3 __attribute__((address_space(3)))

// Async global->LDS (CDNA5, ASYNCcnt-tracked) if the toolchain exposes it;
// otherwise fall back to the register-staged pipeline. (__has_builtin is false
// in the host pass, which harmlessly compiles the fallback there.)
#if __has_builtin(__builtin_amdgcn_global_load_async_to_lds_b128) && \
    __has_builtin(__builtin_amdgcn_global_load_async_to_lds_b32)  && \
    __has_builtin(__builtin_amdgcn_s_wait_asynccnt)
#define USE_ASYNC_LDS 1
#else
#define USE_ASYNC_LDS 0
#endif

static constexpr int Bsz = 8;
static constexpr int Nn  = 2048;
static constexpr int Ff  = 128;

// K index within a 16-bit A fragment (16x32, MxK) for element e (0..15),
// lane group g = lane>>4.  (ISA 7.12.2: VGPR0 K=0,1 / K=8,9; VGPR4 K=16,17 / 24,25)
__device__ __forceinline__ int a_k(int e, int g) {
    return e + 8 * g + ((e >= 8) ? 8 : 0);
}

__device__ __forceinline__ v16h cat8(v8h lo, v8h hi) {
    return __builtin_shufflevector(lo, hi, 0, 1, 2, 3, 4, 5, 6, 7,
                                   8, 9, 10, 11, 12, 13, 14, 15);
}

// ---------------------------------------------------------------------------
// Stage A: WhT[b][o][j] = (h @ W)^T  (f32 in, f16 out), WMMA 16x16x32 f16->f32
// grid = 128 blocks of 256 threads (8 waves); each wave computes 16 rows x 128.
// ---------------------------------------------------------------------------
__global__ __launch_bounds__(256) void wh_gemm_kernel(
    const float* __restrict__ h, const float* __restrict__ W,
    f16* __restrict__ WhT)
{
    // W transposed: [n][k], stride 136 halves (272 B: 16B-aligned, conflict-free)
    __shared__ __align__(16) f16 ldsW[128 * 136];

    const int t = threadIdx.x;
    for (int idx = t; idx < 128 * 128; idx += 256) {
        int k = idx >> 7, n = idx & 127;
        ldsW[n * 136 + k] = (f16)W[idx];           // coalesced read of W[k][n]
    }
    __syncthreads();

    const int wave = t >> 5;
    const int lane = t & 31;
    const int g    = lane >> 4;
    const int m    = lane & 15;
    const int arow = blockIdx.x * 128 + wave * 16 + m;   // A-fragment row (lane = M)

    // Load + convert 4 A fragments (K chunks of 32) directly from global h.
    v16h afrag[4];
    const float* hrow = h + (size_t)arow * Ff;
    #pragma unroll
    for (int kc = 0; kc < 4; ++kc) {
        #pragma unroll
        for (int e = 0; e < 16; e += 2) {
            int K = kc * 32 + a_k(e, g);           // pairs contiguous in k
            afrag[kc][e]     = (f16)hrow[K];
            afrag[kc][e + 1] = (f16)hrow[K + 1];
        }
    }

    const int outrow_base = blockIdx.x * 128 + wave * 16;  // global row block
    const int bb = outrow_base >> 11;                      // batch of this block
    const int jb = outrow_base & 2047;                     // j within batch

    #pragma unroll
    for (int nt = 0; nt < 8; ++nt) {
        // Issue all 8 ds_load_b128 first, then run the 4 WMMAs behind one wait.
        v16h bf[4];
        #pragma unroll
        for (int kc = 0; kc < 4; ++kc) {
            const f16* col = &ldsW[(nt * 16 + m) * 136 + kc * 32 + 8 * g];
            bf[kc] = cat8(*(const v8h*)col, *(const v8h*)(col + 16));
        }
        v8f acc = {0.f, 0.f, 0.f, 0.f, 0.f, 0.f, 0.f, 0.f};
        #pragma unroll
        for (int kc = 0; kc < 4; ++kc)
            acc = __builtin_amdgcn_wmma_f32_16x16x32_f16(
                false, afrag[kc], false, bf[kc], (short)0, acc, false, false);

        // Pack 8 rows (consecutive j) into one 16-byte store.
        v8h pk;
        #pragma unroll
        for (int r = 0; r < 8; ++r) pk[r] = (f16)acc[r];
        *(v8h*)&WhT[((size_t)(bb * Ff + nt * 16 + m) << 11) + jb + 8 * g] = pk;
    }
}

// ---------------------------------------------------------------------------
// src = Wh @ a1, dst = Wh @ a2 from transposed WhT (lane-coalesced reads).
// ---------------------------------------------------------------------------
__global__ __launch_bounds__(256) void srcdst_kernel(
    const f16* __restrict__ WhT, const float* __restrict__ a,
    float* __restrict__ src, float* __restrict__ dst)
{
    int row = blockIdx.x * 256 + threadIdx.x;     // b*2048 + j
    if (row >= Bsz * Nn) return;
    int b = row >> 11, j = row & 2047;
    const f16* base = WhT + ((size_t)b * Ff << 11) + j;
    float s = 0.f, d = 0.f;
    #pragma unroll 8
    for (int o = 0; o < Ff; ++o) {
        float v = (float)base[(size_t)o << 11];   // coalesced across lanes
        s += v * a[o];
        d += v * a[Ff + o];
    }
    src[row] = s;
    dst[row] = d;
}

// ---------------------------------------------------------------------------
// Stage B: fused masked-softmax attention, flash style, double-buffered.
// grid = 8 batches * 32 row-chunks = 256 blocks, 128 threads (4 waves).
// ---------------------------------------------------------------------------
__global__ __launch_bounds__(128) void attn_kernel(
    const f16* __restrict__ WhT, const int* __restrict__ adj,
    const float* __restrict__ src, const float* __restrict__ dst,
    float* __restrict__ out)
{
    // [n][j] tile, stride 40 halves (80 B: 16B-aligned, conflict-free), x2 buffers
    __shared__ __align__(16) f16 ldsWhT[2][128 * 40];
    __shared__ float ldsDst[2][32];

    const int t    = threadIdx.x;
    const int wave = t >> 5;
    const int lane = t & 31;
    const int g    = lane >> 4;
    const int m    = lane & 15;
    const int b      = blockIdx.x >> 5;          // batch
    const int ichunk = blockIdx.x & 31;          // 64-row chunk
    const int i0     = ichunk * 64 + wave * 16;  // wave's row block
    const int myrow  = i0 + m;                   // this lane's row (A-frag row)
    const int bN     = b * Nn;
    constexpr int NCH = Nn / 32;

    const float srcv   = src[bN + myrow];
    const int*  adjrow = adj + (size_t)(bN + myrow) * Nn;
    const f16*  whb    = WhT + ((size_t)b * Ff << 11);   // this batch's WhT

    float m_run = -INFINITY;
    float l_run = 0.f;
    v8f acc[8];
    #pragma unroll
    for (int nt = 0; nt < 8; ++nt)
        acc[nt] = (v8f){0.f, 0.f, 0.f, 0.f, 0.f, 0.f, 0.f, 0.f};

#if USE_ASYNC_LDS
    // ---- async global->LDS pipeline (no VGPR staging, ASYNCcnt-tracked) ----
    auto issue_tile = [&](int buf, int j0n) {
        const f16* gp = whb + ((size_t)t << 11) + j0n;   // row n = t, 64 bytes
        f16*       lp = &ldsWhT[buf][t * 40];
        #pragma unroll
        for (int q = 0; q < 4; ++q)
            __builtin_amdgcn_global_load_async_to_lds_b128(
                (AS1 v4i*)(gp + 8 * q), (AS3 v4i*)(lp + 8 * q), 0, 0);
        if (t < 32)
            __builtin_amdgcn_global_load_async_to_lds_b32(
                (AS1 int*)(dst + bN + j0n + t), (AS3 int*)&ldsDst[buf][t], 0, 0);
    };
    issue_tile(0, 0);
    __builtin_amdgcn_s_wait_asynccnt(0);
    __syncthreads();
#else
    // ---- fallback: global -> regs -> (compute) -> LDS pipeline ----
    float4 rbuf[4];
    float  rdst;
    auto issue_loads = [&](int j0n) {
        const float4* p4 = (const float4*)(whb + ((size_t)t << 11) + j0n);
        rbuf[0] = p4[0]; rbuf[1] = p4[1]; rbuf[2] = p4[2]; rbuf[3] = p4[3];
        rdst = (t < 32) ? dst[bN + j0n + t] : 0.f;
    };
    auto commit = [&](int buf) {
        float4* q4 = (float4*)&ldsWhT[buf][t * 40];
        q4[0] = rbuf[0]; q4[1] = rbuf[1]; q4[2] = rbuf[2]; q4[3] = rbuf[3];
        if (t < 32) ldsDst[buf][t] = rdst;
    };
    issue_loads(0);
    commit(0);
    __syncthreads();
#endif

    for (int jc = 0; jc < NCH; ++jc) {
        const int j0  = jc * 32;
        const int cur = jc & 1;

#if USE_ASYNC_LDS
        if (jc + 1 < NCH) issue_tile(cur ^ 1, j0 + 32);  // overlap with compute
#else
        if (jc + 1 < NCH) issue_loads(j0 + 32);
#endif

        // Prefetch next chunk's adj cachelines (global_prefetch_b8).
        if (jc + 1 < NCH) {
            __builtin_prefetch(adjrow + j0 + 32, 0, 1);
            __builtin_prefetch(adjrow + j0 + 48, 0, 1);
        }

        // e values in A-fragment element order; adj read as int2 pairs.
        float ev[16];
        float cmax = -INFINITY;
        #pragma unroll
        for (int e = 0; e < 16; e += 2) {
            int K = a_k(e, g);
            int2 aj = *(const int2*)(adjrow + j0 + K);
            float x0 = srcv + ldsDst[cur][K];
            float x1 = srcv + ldsDst[cur][K + 1];
            float l0 = x0 > 0.f ? x0 : LRELU_SLOPE * x0;
            float l1 = x1 > 0.f ? x1 : LRELU_SLOPE * x1;
            ev[e]     = (aj.x > 0) ? l0 : NEG_BIG;
            ev[e + 1] = (aj.y > 0) ? l1 : NEG_BIG;
            cmax = fmaxf(cmax, fmaxf(ev[e], ev[e + 1]));
        }
        // lanes L and L+16 share row m: combine their half-chunks
        cmax = fmaxf(cmax, __shfl_xor(cmax, 16, 32));
        float m_new = fmaxf(m_run, cmax);
        float scale = __expf(m_run - m_new);   // exp(-inf)=0 on first chunk

        float psum = 0.f;
        v16h pfrag;
        #pragma unroll
        for (int e = 0; e < 16; ++e) {
            float p = __expf(ev[e] - m_new);
            psum += p;
            pfrag[e] = (f16)p;
        }
        psum += __shfl_xor(psum, 16, 32);
        l_run = l_run * scale + psum;
        m_run = m_new;

        // Rescale accumulators: acc VGPR r belongs to row r + 8*g; row-m stats
        // live in lanes m and m+16, so source lane r+8*g holds row r+8*g.
        float srow[8];
        #pragma unroll
        for (int r = 0; r < 8; ++r) srow[r] = __shfl(scale, r + 8 * g, 32);
        #pragma unroll
        for (int nt = 0; nt < 8; ++nt)
            #pragma unroll
            for (int r = 0; r < 8; ++r) acc[nt][r] *= srow[r];

        // acc += P(16x32) * Wh(32x128): two groups of 4, loads issued ahead of
        // the WMMAs so ds latency hides behind matrix work.
        #pragma unroll
        for (int half = 0; half < 2; ++half) {
            v16h bf[4];
            #pragma unroll
            for (int q = 0; q < 4; ++q) {
                int nt = half * 4 + q;
                const f16* col = &ldsWhT[cur][(nt * 16 + m) * 40 + 8 * g];
                bf[q] = cat8(*(const v8h*)col, *(const v8h*)(col + 16));
            }
            #pragma unroll
            for (int q = 0; q < 4; ++q)
                acc[half * 4 + q] = __builtin_amdgcn_wmma_f32_16x16x32_f16(
                    false, pfrag, false, bf[q], (short)0, acc[half * 4 + q],
                    false, false);
        }

#if USE_ASYNC_LDS
        if (jc + 1 < NCH) __builtin_amdgcn_s_wait_asynccnt(0);
        __syncthreads();
#else
        if (jc + 1 < NCH) commit(cur ^ 1);
        __syncthreads();
#endif
    }

    // Normalize by running softmax denominator and store f32 output.
    float lrow[8];
    #pragma unroll
    for (int r = 0; r < 8; ++r) {
        float l = __shfl(l_run, r + 8 * g, 32);
        lrow[r] = 1.0f / l;
    }
    #pragma unroll
    for (int nt = 0; nt < 8; ++nt) {
        #pragma unroll
        for (int r = 0; r < 8; ++r) {
            int orow = i0 + r + 8 * g;
            out[(size_t)(bN + orow) * Ff + nt * 16 + m] = acc[nt][r] * lrow[r];
        }
    }
}

// ---------------------------------------------------------------------------
extern "C" void kernel_launch(void* const* d_in, const int* in_sizes, int n_in,
                              void* d_out, int out_size, void* d_ws, size_t ws_size,
                              hipStream_t stream) {
    const float* h   = (const float*)d_in[0];   // (8,2048,128) f32
    const int*   adj = (const int*)d_in[1];     // (8,2048,2048) i32
    const float* W   = (const float*)d_in[2];   // (128,128) f32
    const float* a   = (const float*)d_in[3];   // (256,) f32
    float* out = (float*)d_out;                 // (8,2048,128) f32

    // workspace layout: WhT f16 (4 MB) | src f32 (64 KB) | dst f32 (64 KB)
    char*  ws   = (char*)d_ws;
    f16*   WhT  = (f16*)ws;
    float* src  = (float*)(ws + (size_t)Bsz * Nn * Ff * sizeof(f16));
    float* dst  = src + Bsz * Nn;

    wh_gemm_kernel<<<dim3(128), dim3(256), 0, stream>>>(h, W, WhT);
    srcdst_kernel<<<dim3(64), dim3(256), 0, stream>>>(WhT, a, src, dst);
    attn_kernel<<<dim3(256), dim3(128), 0, stream>>>(WhT, adj, src, dst, out);
}
